// LMMD_loss_23081154248884
// MI455X (gfx1250) — matrix operand order repaired and checked by reading, not compiled, not run on hardware
//
#include <hip/hip_runtime.h>
#include <hip/hip_bf16.h>

#define B_ROWS 4096
#define N_ROWS 8192
#define DIM    256
#define NCLS   31

// workspace layout (float offsets)
#define OFF_SCNT  0     // [32] source class counts
#define OFF_TCNT  32    // [32] target class prob sums
#define OFF_TPRES 64    // [32] target argmax presence counts
#define OFF_INVS  96    // [32] mask_c / safe(s_cnt_c)
#define OFF_INVT  128   // [32] mask_c / safe(t_cnt_c)
#define OFF_SCAL  160   // [0]=1/n_idx [1]=sum(sq) [2]=loss acc [3..7]=inv_bw[5]
#define OFF_MSUM  176   // [256] column sums of `total`
#define OFF_SQ    432   // [8192] row squared norms
#define WS_FLOATS (OFF_SQ + N_ROWS)

typedef __attribute__((ext_vector_type(2)))  __bf16 v2bf;
typedef __attribute__((ext_vector_type(8)))  __bf16 v8bf;
typedef __attribute__((ext_vector_type(16))) __bf16 v16bf;
typedef __attribute__((ext_vector_type(8)))  float  v8f;

static __device__ inline v8f wmma_bf16(v16bf a, v16bf b, v8f c) {
  // D = A(16x32 bf16) * B(32x16 bf16) + C(16x16 f32)
  return __builtin_amdgcn_wmma_f32_16x16x32_bf16(false, a, false, b,
                                                 (short)0, c, false, false);
}

#define LDSTR 40  // 32 k-values + 8 pad (bf16); row stride 80B = multiple of 16B

// A fragment 16x32 bf16: lane m=L&15, half=L>>4;
// comps 0..7 -> K = half*8 + c ; comps 8..15 -> K = 16 + half*8 + c
// Both chunks are 16B-aligned -> two ds_load_b128.
static __device__ inline v16bf load_a(const __bf16* base, int rowbase, int lane) {
  const int m = lane & 15, half = lane >> 4;
  const __bf16* p = base + (size_t)(rowbase + m) * LDSTR + half * 8;
  v8bf c0 = *(const v8bf*)(p);        // K = half*8 + 0..7
  v8bf c1 = *(const v8bf*)(p + 16);   // K = 16 + half*8 + 0..7
  return __builtin_shufflevector(c0, c1, 0, 1, 2, 3, 4, 5, 6, 7,
                                         8, 9, 10, 11, 12, 13, 14, 15);
}

// B fragment 32x16 bf16: lane n=L&15, khalf=L>>4;
// comps 0..7 -> K = khalf*8 + c ; comps 8..15 -> K = 16 + khalf*8 + c
// Both chunks are 16B-aligned -> two ds_load_b128.
static __device__ inline v16bf load_b(const __bf16* base, int colbase, int lane) {
  const int n = lane & 15, kh = lane >> 4;
  const __bf16* p = base + (size_t)(colbase + n) * LDSTR + kh * 8;
  v8bf c0 = *(const v8bf*)(p);
  v8bf c1 = *(const v8bf*)(p + 16);
  return __builtin_shufflevector(c0, c1, 0, 1, 2, 3, 4, 5, 6, 7,
                                         8, 9, 10, 11, 12, 13, 14, 15);
}

static __device__ inline v2bf split_hi(float x0, float x1) {
  v2bf h; h[0] = (__bf16)x0; h[1] = (__bf16)x1; return h;
}
static __device__ inline v2bf split_lo(float x0, float x1, v2bf h) {
  v2bf l; l[0] = (__bf16)(x0 - (float)h[0]); l[1] = (__bf16)(x1 - (float)h[1]); return l;
}

// ---------------- phase-0 / phase-1 kernels ----------------

__global__ void k_init(float* ws) {
  int t = threadIdx.x;
  if (t < OFF_SQ) ws[t] = 0.f;  // zero accumulated stats (sq region fully rewritten)
}

// one wave32 per row: sq[i] = ||x_i||^2 ; also accumulate sum(sq)
__global__ __launch_bounds__(256) void k_rowstats(const float* __restrict__ src,
                                                  const float* __restrict__ tgt,
                                                  float* __restrict__ ws) {
  const int row  = blockIdx.x * 8 + (threadIdx.x >> 5);
  const int lane = threadIdx.x & 31;
  const float* p = (row < B_ROWS) ? src + (size_t)row * DIM
                                  : tgt + (size_t)(row - B_ROWS) * DIM;
  float s = 0.f;
#pragma unroll
  for (int e = 0; e < 8; e++) { float x = p[lane + 32 * e]; s = fmaf(x, x, s); }
#pragma unroll
  for (int off = 16; off > 0; off >>= 1) s += __shfl_xor(s, off, 32);
  if (lane == 0) {
    ws[OFF_SQ + row] = s;
    atomicAdd(&ws[OFF_SCAL + 1], s);
  }
}

// column sums of `total` (closed-form sum(l2) = 2N*sum(sq) - 2*||colsum||^2)
__global__ void k_colsum(const float* __restrict__ src, const float* __restrict__ tgt,
                         float* __restrict__ ws) {
  const int col = threadIdx.x;
  const int r0  = blockIdx.x * 256;
  float s = 0.f;
  for (int r = r0; r < r0 + 256; r++)
    s += (r < B_ROWS) ? src[(size_t)r * DIM + col]
                      : tgt[(size_t)(r - B_ROWS) * DIM + col];
  atomicAdd(&ws[OFF_MSUM + col], s);
}

// class stats: s_cnt, t_cnt, t-present (per-row argmax), LDS pre-reduced
__global__ void k_labels(const int* __restrict__ slab, const float* __restrict__ tlab,
                         float* __restrict__ ws) {
  __shared__ float ls[32], lt[32], lp[32];
  const int tid = threadIdx.x;
  if (tid < 32) { ls[tid] = 0.f; lt[tid] = 0.f; lp[tid] = 0.f; }
  __syncthreads();
  const int i = blockIdx.x * 256 + tid;
  atomicAdd(&ls[slab[i]], 1.f);
  float best = -1.f; int bc = 0;
  for (int c = 0; c < NCLS; c++) {
    float v = tlab[(size_t)i * NCLS + c];
    atomicAdd(&lt[c], v);
    if (v > best) { best = v; bc = c; }  // first-max, matches jnp.argmax
  }
  atomicAdd(&lp[bc], 1.f);
  __syncthreads();
  if (tid < 32) {
    atomicAdd(&ws[OFF_SCNT + tid],  ls[tid]);
    atomicAdd(&ws[OFF_TCNT + tid],  lt[tid]);
    atomicAdd(&ws[OFF_TPRES + tid], lp[tid]);
  }
}

// mask, inv class scales, n_idx, closed-form sum(l2) -> bandwidths
__global__ __launch_bounds__(256) void k_finalize(float* __restrict__ ws) {
  __shared__ float red[256];
  const int tid = threadIdx.x;

  float maskv = 0.f;
  if (tid < 32) {
    float sc = ws[OFF_SCNT + tid];
    float tc = ws[OFF_TCNT + tid];
    float pr = ws[OFF_TPRES + tid];
    maskv = (tid < NCLS && sc > 0.f && pr > 0.f) ? 1.f : 0.f;
    ws[OFF_INVS + tid] = maskv / ((sc == 0.f) ? 100.f : sc);
    ws[OFF_INVT + tid] = maskv / ((tc == 0.f) ? 100.f : tc);
  }
  red[tid] = maskv;
  __syncthreads();
  for (int s = 128; s > 0; s >>= 1) { if (tid < s) red[tid] += red[tid + s]; __syncthreads(); }
  const float n_idx = fmaxf(red[0], 1.f);
  __syncthreads();

  float m = ws[OFF_MSUM + tid];
  red[tid] = m * m;
  __syncthreads();
  for (int s = 128; s > 0; s >>= 1) { if (tid < s) red[tid] += red[tid + s]; __syncthreads(); }

  if (tid == 0) {
    const float sumsq  = ws[OFF_SCAL + 1];
    const float sum_l2 = 2.f * (float)N_ROWS * sumsq - 2.f * red[0];
    const float denom  = (float)((long long)N_ROWS * N_ROWS - N_ROWS);
    float bw = fmaxf(sum_l2 / denom, 1e-6f) * 0.25f;  // / KERNEL_MUL^(5//2)
    float mul = 1.f;
    for (int q = 0; q < 5; q++) { ws[OFF_SCAL + 3 + q] = 1.f / fmaxf(bw * mul, 1e-6f); mul *= 2.f; }
    ws[OFF_SCAL + 0] = 1.f / n_idx;
    ws[OFF_SCAL + 2] = 0.f;  // loss accumulator
  }
}

// per-row class vector entry (u_i / v_j): never materialize the 4096^2 weights
static __device__ inline float class_val(int g, int c, const int* __restrict__ slab,
                                         const float* __restrict__ tlab,
                                         const float* __restrict__ ws) {
  if (c >= NCLS) return 0.f;
  if (g < B_ROWS) return (slab[g] == c) ? ws[OFF_INVS + c] : 0.f;
  return tlab[(size_t)(g - B_ROWS) * NCLS + c] * ws[OFF_INVT + c];
}

// ---------------- fused WMMA tile kernel ----------------
// 128x128 tile of the 8192x8192 kernel matrix per workgroup.
// Gram tile: 8 k-steps x 3 bf16 WMMAs (hi/lo split, ~fp32 accuracy).
// Weight tile: rank-31 u.v outer product, 1 k-step x 3 WMMAs.
__global__ __launch_bounds__(256) void k_main(const float* __restrict__ src,
                                              const float* __restrict__ tgt,
                                              const int*   __restrict__ slab,
                                              const float* __restrict__ tlab,
                                              float* __restrict__ ws) {
  const int bi = blockIdx.y, bj = blockIdx.x;
  const bool irow_s = (bi < 32), jcol_s = (bj < 32);
  float coef;
  if (irow_s == jcol_s)      coef = 1.f;   // ss / tt blocks
  else if (irow_s)           coef = -2.f;  // st block
  else                       return;       // ts block has zero weight: skip

  __shared__ __attribute__((aligned(16))) __bf16 sA_hi[128 * LDSTR];
  __shared__ __attribute__((aligned(16))) __bf16 sA_lo[128 * LDSTR];
  __shared__ __attribute__((aligned(16))) __bf16 sB_hi[128 * LDSTR];
  __shared__ __attribute__((aligned(16))) __bf16 sB_lo[128 * LDSTR];
  __shared__ float s_sqr[128], s_sqc[128];
  __shared__ float s_red[256];

  const int tid  = threadIdx.x;
  const int lane = tid & 31;
  const int w    = tid >> 5;
  const int wr   = (w & 3) * 32;   // wave sub-tile: 32 rows x 64 cols
  const int wc   = (w >> 2) * 64;
  const int i0   = bi * 128, j0 = bj * 128;

  if (tid < 128) s_sqr[tid] = ws[OFF_SQ + i0 + tid];
  else           s_sqc[tid - 128] = ws[OFF_SQ + j0 + (tid - 128)];

  v8f acc[2][4];
#pragma unroll
  for (int fr = 0; fr < 2; fr++)
#pragma unroll
    for (int fc = 0; fc < 4; fc++)
#pragma unroll
      for (int e = 0; e < 8; e++) acc[fr][fc][e] = 0.f;

  for (int k0 = 0; k0 < DIM; k0 += 32) {
    // stage 128x32 fp32 -> bf16 hi/lo (float2 loads, packed 4B LDS stores)
#pragma unroll
    for (int e = 0; e < 8; e++) {
      int idx = tid + 256 * e;          // 0..2047 pair slots
      int r   = idx >> 4;               // row 0..127
      int kp  = (idx & 15) * 2;         // even k offset 0..30
      {
        int gi = i0 + r;
        const float* p = (gi < B_ROWS) ? (src + (size_t)gi * DIM + k0 + kp)
                                       : (tgt + (size_t)(gi - B_ROWS) * DIM + k0 + kp);
        float2 x = *(const float2*)p;
        if (k0 + 32 < DIM) __builtin_prefetch(p + 32, 0, 0);
        v2bf h = split_hi(x.x, x.y);
        *(v2bf*)&sA_hi[r * LDSTR + kp] = h;
        *(v2bf*)&sA_lo[r * LDSTR + kp] = split_lo(x.x, x.y, h);
      }
      {
        int gj = j0 + r;
        const float* p = (gj < B_ROWS) ? (src + (size_t)gj * DIM + k0 + kp)
                                       : (tgt + (size_t)(gj - B_ROWS) * DIM + k0 + kp);
        float2 x = *(const float2*)p;
        v2bf h = split_hi(x.x, x.y);
        *(v2bf*)&sB_hi[r * LDSTR + kp] = h;
        *(v2bf*)&sB_lo[r * LDSTR + kp] = split_lo(x.x, x.y, h);
      }
    }
    __syncthreads();

    v16bf ah[2], al[2];
#pragma unroll
    for (int fr = 0; fr < 2; fr++) {
      ah[fr] = load_a(sA_hi, wr + fr * 16, lane);
      al[fr] = load_a(sA_lo, wr + fr * 16, lane);
    }
#pragma unroll
    for (int fc = 0; fc < 4; fc++) {
      v16bf bh = load_b(sB_hi, wc + fc * 16, lane);
      v16bf bl = load_b(sB_lo, wc + fc * 16, lane);
#pragma unroll
      for (int fr = 0; fr < 2; fr++) {
        acc[fr][fc] = wmma_bf16(ah[fr], bh, acc[fr][fc]);  // hi*hi
        acc[fr][fc] = wmma_bf16(ah[fr], bl, acc[fr][fc]);  // hi*lo
        acc[fr][fc] = wmma_bf16(al[fr], bh, acc[fr][fc]);  // lo*hi
      }
    }
    __syncthreads();
  }

  // ---- weight tile: u_i (rows) and v_j (cols), 31 classes padded to 32 ----
#pragma unroll
  for (int e = 0; e < 8; e++) {
    int idx = tid + 256 * e;
    int r   = idx >> 4;
    int c0  = (idx & 15) * 2;
    int gi = i0 + r, gj = j0 + r;
    float u0 = class_val(gi, c0, slab, tlab, ws);
    float u1 = class_val(gi, c0 + 1, slab, tlab, ws);
    float v0 = class_val(gj, c0, slab, tlab, ws);
    float v1 = class_val(gj, c0 + 1, slab, tlab, ws);
    v2bf uh = split_hi(u0, u1);
    *(v2bf*)&sA_hi[r * LDSTR + c0] = uh;
    *(v2bf*)&sA_lo[r * LDSTR + c0] = split_lo(u0, u1, uh);
    v2bf vh = split_hi(v0, v1);
    *(v2bf*)&sB_hi[r * LDSTR + c0] = vh;
    *(v2bf*)&sB_lo[r * LDSTR + c0] = split_lo(v0, v1, vh);
  }
  __syncthreads();

  v8f wacc[2][4];
#pragma unroll
  for (int fr = 0; fr < 2; fr++)
#pragma unroll
    for (int fc = 0; fc < 4; fc++)
#pragma unroll
      for (int e = 0; e < 8; e++) wacc[fr][fc][e] = 0.f;

  {
    v16bf uh[2], ul[2];
#pragma unroll
    for (int fr = 0; fr < 2; fr++) {
      uh[fr] = load_a(sA_hi, wr + fr * 16, lane);
      ul[fr] = load_a(sA_lo, wr + fr * 16, lane);
    }
#pragma unroll
    for (int fc = 0; fc < 4; fc++) {
      v16bf vh = load_b(sB_hi, wc + fc * 16, lane);
      v16bf vl = load_b(sB_lo, wc + fc * 16, lane);
#pragma unroll
      for (int fr = 0; fr < 2; fr++) {
        wacc[fr][fc] = wmma_bf16(uh[fr], vh, wacc[fr][fc]);
        wacc[fr][fc] = wmma_bf16(uh[fr], vl, wacc[fr][fc]);
        wacc[fr][fc] = wmma_bf16(ul[fr], vh, wacc[fr][fc]);
      }
    }
  }

  // ---- fused epilogue: l2 -> 5x exp -> weighted partial sum ----
  const float ib0 = ws[OFF_SCAL + 3], ib1 = ws[OFF_SCAL + 4], ib2 = ws[OFF_SCAL + 5],
              ib3 = ws[OFF_SCAL + 6], ib4 = ws[OFF_SCAL + 7];
  const int half = lane >> 4, n15 = lane & 15;
  float part = 0.f;
#pragma unroll
  for (int fr = 0; fr < 2; fr++) {
#pragma unroll
    for (int fc = 0; fc < 4; fc++) {
      const float sqj = s_sqc[wc + fc * 16 + n15];
#pragma unroll
      for (int v = 0; v < 8; v++) {
        const float sqi = s_sqr[wr + fr * 16 + half * 8 + v];
        const float g   = acc[fr][fc][v];
        const float l2  = fmaxf(sqi + sqj - 2.f * g, 0.f);
        const float ks  = __expf(-l2 * ib0) + __expf(-l2 * ib1) + __expf(-l2 * ib2)
                        + __expf(-l2 * ib3) + __expf(-l2 * ib4);
        part += wacc[fr][fc][v] * ks;
      }
    }
  }
  part *= coef;

  s_red[tid] = part;
  __syncthreads();
  for (int s = 128; s > 0; s >>= 1) { if (tid < s) s_red[tid] += s_red[tid + s]; __syncthreads(); }
  if (tid == 0) atomicAdd(&ws[OFF_SCAL + 2], s_red[0]);
}

__global__ void k_out(const float* __restrict__ ws, float* __restrict__ out) {
  out[0] = ws[OFF_SCAL + 2] * ws[OFF_SCAL + 0];  // / n_idx
}

extern "C" void kernel_launch(void* const* d_in, const int* in_sizes, int n_in,
                              void* d_out, int out_size, void* d_ws, size_t ws_size,
                              hipStream_t stream) {
  (void)in_sizes; (void)n_in; (void)out_size; (void)ws_size;
  const float* src = (const float*)d_in[0];
  const float* tgt = (const float*)d_in[1];
  const int*   slb = (const int*)d_in[2];
  const float* tlb = (const float*)d_in[3];
  float* out = (float*)d_out;
  float* ws  = (float*)d_ws;

  k_init<<<1, 512, 0, stream>>>(ws);
  k_rowstats<<<N_ROWS / 8, 256, 0, stream>>>(src, tgt, ws);
  k_colsum<<<N_ROWS / 256, 256, 0, stream>>>(src, tgt, ws);
  k_labels<<<B_ROWS / 256, 256, 0, stream>>>(slb, tlb, ws);
  k_finalize<<<1, 256, 0, stream>>>(ws);
  dim3 grid(64, 64);
  k_main<<<grid, 256, 0, stream>>>(src, tgt, slb, tlb, ws);
  k_out<<<1, 1, 0, stream>>>(ws, out);
}
